// TSP_Encoder_56487409877516
// MI455X (gfx1250) — matrix operand order repaired and checked by reading, not compiled, not run on hardware
//
#include <hip/hip_runtime.h>

// ---------------- problem constants ----------------
#define Bc 64
#define Nc 500
#define Np 512   // N padded for transposed ek/ekv storage
#define Dc 128
#define Fc 512
#define Lc 6
#define EPSc 1e-5f
#define RT 32    // ceil(N/16) row tiles

// ---------------- types ----------------
typedef __attribute__((ext_vector_type(16))) __bf16 v16bf;
typedef __attribute__((ext_vector_type(8)))  float  v8f;

#define WMMA_BF16(a, b, c) \
  __builtin_amdgcn_wmma_f32_16x16x32_bf16(false, (a), false, (b), (short)0, (c), false, false)

// ---------------- fragment loaders (contiguous k = 16*half + s, wave32) ----------------
// A: 16x32 bf16 (MxK) from LDS. lane 0-15 -> rows, half selects K groups.
__device__ __forceinline__ void load_a_frag(const __bf16* smem, int ldm, v16bf& a) {
  int lane = threadIdx.x & 31;
  int row  = lane & 15;
  int half = lane >> 4;
#pragma unroll
  for (int i = 0; i < 8; ++i) {
    int k0 = half * 16 + i * 2;
    a[2 * i]     = smem[row * ldm + k0];
    a[2 * i + 1] = smem[row * ldm + k0 + 1];
  }
}

// B: 32x16 (KxN). Lane (col = lane&15, kb = (lane>>4)*16) needs B[kb..kb+15][col].
// With K contiguous in memory per output column this is one 32-byte vector load.
__device__ __forceinline__ v16bf load_b_frag_kcontig(const __bf16* p) {
  return *(const v16bf*)p;  // 32B aligned by construction -> 2x global_load_b128
}

// ---------------- kernels ----------------
__global__ void k_cvt(const float* __restrict__ src, __bf16* __restrict__ dst, long n) {
  long i = (long)blockIdx.x * blockDim.x + threadIdx.x;
  if (i < n) dst[i] = (__bf16)src[i];
}

__global__ void k_embed(const float* __restrict__ data, const float* __restrict__ W_emb,
                        const float* __restrict__ b_emb, float* __restrict__ x,
                        __bf16* __restrict__ xb) {
  long idx = (long)blockIdx.x * blockDim.x + threadIdx.x;
  if (idx >= (long)Bc * Nc * Dc) return;
  int d  = (int)(idx % Dc);
  long bn = idx / Dc;
  float v = data[bn * 2 + 0] * W_emb[d * 2 + 0] + data[bn * 2 + 1] * W_emb[d * 2 + 1] + b_emb[d];
  x[idx]  = v;
  xb[idx] = (__bf16)v;
}

__global__ void k_reembed(const float* __restrict__ W_out, const float* __restrict__ b_out,
                          float* __restrict__ x, __bf16* __restrict__ xb) {
  __shared__ float row[Dc];
  int b = blockIdx.x;
  int n = (blockIdx.y == 0) ? 0 : (Nc - 1);
  int d = threadIdx.x;
  long base = ((long)b * Nc + n) * Dc;
  row[d] = x[base + d];
  __syncthreads();
  float acc = b_out[d];
  const float* w = W_out + (long)d * Dc;
#pragma unroll 8
  for (int i = 0; i < Dc; ++i) acc += row[i] * w[i];
  x[base + d]  = acc;
  xb[base + d] = (__bf16)acc;
}

// q,k,v GEMMs + epilogue: sq = sigmoid(q) f32; ekT/ekvT bf16 transposed [B, D, Np]
__global__ void k_qkv(const __bf16* __restrict__ xb, const __bf16* __restrict__ Wqb,
                      const __bf16* __restrict__ Wkb, const __bf16* __restrict__ Wvb,
                      float* __restrict__ sq, __bf16* __restrict__ ekT,
                      __bf16* __restrict__ ekvT) {
  __shared__ __bf16 As[16 * Dc];
  int b  = blockIdx.x;
  int r0 = blockIdx.y * 16;
  for (int i = threadIdx.x; i < 16 * Dc; i += 256) {
    int r = i >> 7, c = i & (Dc - 1);
    int rr = r0 + r; if (rr >= Nc) rr = Nc - 1;
    As[i] = xb[((long)b * Nc + rr) * Dc + c];
  }
  __syncthreads();
  int lane = threadIdx.x & 31;
  int col  = lane & 15;
  int kb   = (lane >> 4) * 16;
  int d0   = (threadIdx.x >> 5) * 16;
  v8f cq = {}, ck = {}, cv = {};
#pragma unroll
  for (int ks = 0; ks < Dc / 32; ++ks) {
    v16bf a; load_a_frag(As + ks * 32, Dc, a);
    long wo = (long)(d0 + col) * Dc + ks * 32 + kb;
    v16bf bq = load_b_frag_kcontig(Wqb + wo);
    v16bf bk = load_b_frag_kcontig(Wkb + wo);
    v16bf bv = load_b_frag_kcontig(Wvb + wo);
    cq = WMMA_BF16(a, bq, cq);
    ck = WMMA_BF16(a, bk, ck);
    cv = WMMA_BF16(a, bv, cv);
  }
  int mb = (lane >> 4) * 8, n = lane & 15;
#pragma unroll
  for (int j = 0; j < 8; ++j) {
    int r = r0 + mb + j;
    long ot = ((long)b * Dc + d0 + n) * Np + r;   // transposed
    if (r < Nc) {
      float e = __expf(ck[j]);
      sq[((long)b * Nc + r) * Dc + d0 + n] = 1.0f / (1.0f + __expf(-cq[j]));
      ekT[ot]  = (__bf16)e;
      ekvT[ot] = (__bf16)(e * cv[j]);
    } else {                                      // zero the pad rows [Nc, Np)
      ekT[ot]  = (__bf16)0.0f;
      ekvT[ot] = (__bf16)0.0f;
    }
  }
}

// AFT attention: num = s@ekv, den = s@ek with s = exp(-cfac*dist); y = x + sigmoid(q)*num/den
__global__ void k_attn(const float* __restrict__ dist, const float* __restrict__ log_scale,
                       const float* __restrict__ alpha_l, const float* __restrict__ x,
                       const float* __restrict__ sq, const __bf16* __restrict__ ekT,
                       const __bf16* __restrict__ ekvT, float* __restrict__ y) {
  __shared__ __bf16 Ss[16 * 32];
  int b  = blockIdx.x;
  int r0 = blockIdx.y * 16;
  float cfac = log_scale[0] * alpha_l[0];
  int lane = threadIdx.x & 31;
  int col  = lane & 15;
  int kb   = (lane >> 4) * 16;
  int d0   = (threadIdx.x >> 5) * 16;
  v8f cnum = {}, cden = {};
  const float* distb = dist + (long)b * Nc * Nc;
  const __bf16* pek  = ekT  + ((long)b * Dc + d0 + col) * Np + kb;
  const __bf16* pekv = ekvT + ((long)b * Dc + d0 + col) * Np + kb;
  for (int m0 = 0; m0 < Nc; m0 += 32) {
    for (int i = threadIdx.x; i < 16 * 32; i += 256) {
      int r = i >> 5, c = i & 31;
      int gr = r0 + r, gc = m0 + c;
      float s = 0.f;
      if (gr < Nc && gc < Nc) {
        s = __expf(-cfac * distb[(long)gr * Nc + gc]);
        if (gc + 32 < Nc)
          __builtin_prefetch(&distb[(long)gr * Nc + gc + 32], 0, 1);  // global_prefetch_b8
      }
      Ss[i] = (__bf16)s;
    }
    __syncthreads();
    v16bf a; load_a_frag(Ss, 32, a);
    v16bf bden = load_b_frag_kcontig(pek  + m0);  // pad rows are zero, no clamp needed
    v16bf bnum = load_b_frag_kcontig(pekv + m0);
    cden = WMMA_BF16(a, bden, cden);
    cnum = WMMA_BF16(a, bnum, cnum);
    __syncthreads();
  }
  int mb = (lane >> 4) * 8, n = lane & 15;
#pragma unroll
  for (int j = 0; j < 8; ++j) {
    int r = r0 + mb + j;
    if (r < Nc) {
      long o = ((long)b * Nc + r) * Dc + d0 + n;
      y[o] = x[o] + sq[o] * (cnum[j] / cden[j]);
    }
  }
}

// InstanceNorm over N per (b,d); writes f32 + bf16 copies
__global__ void k_inorm(const float* __restrict__ yin, const float* __restrict__ gamma,
                        const float* __restrict__ beta, float* __restrict__ xout,
                        __bf16* __restrict__ xbout) {
  int b = blockIdx.x, d = threadIdx.x;
  const float* p = yin + (long)b * Nc * Dc + d;
  float s = 0.f, ss = 0.f;
  for (int n = 0; n < Nc; ++n) { float v = p[(long)n * Dc]; s += v; ss += v * v; }
  float m    = s / (float)Nc;
  float var  = ss / (float)Nc - m * m;
  float rstd = rsqrtf(var + EPSc);
  float g = gamma[d], be = beta[d];
  float*  q  = xout  + (long)b * Nc * Dc + d;
  __bf16* qb = xbout + (long)b * Nc * Dc + d;
  for (int n = 0; n < Nc; ++n) {
    float v = (p[(long)n * Dc] - m) * rstd * g + be;
    q[(long)n * Dc]  = v;
    qb[(long)n * Dc] = (__bf16)v;
  }
}

// h = relu(x1 @ W1.T + b1), bf16 out
__global__ void k_ffn1(const __bf16* __restrict__ x1b, const __bf16* __restrict__ W1b,
                       const float* __restrict__ b1, __bf16* __restrict__ h) {
  __shared__ __bf16 As[16 * Dc];
  int b  = blockIdx.x;
  int r0 = blockIdx.y * 16;
  int lane = threadIdx.x & 31;
  int col  = lane & 15;
  int kb   = (lane >> 4) * 16;
  int f0 = blockIdx.z * 128 + (threadIdx.x >> 5) * 16;
  for (int i = threadIdx.x; i < 16 * Dc; i += 256) {
    int r = i >> 7, c = i & (Dc - 1);
    int rr = r0 + r; if (rr >= Nc) rr = Nc - 1;
    As[i] = x1b[((long)b * Nc + rr) * Dc + c];
  }
  __syncthreads();
  v8f acc = {};
#pragma unroll
  for (int ks = 0; ks < Dc / 32; ++ks) {
    v16bf a; load_a_frag(As + ks * 32, Dc, a);
    v16bf bw = load_b_frag_kcontig(W1b + (long)(f0 + col) * Dc + ks * 32 + kb);
    acc = WMMA_BF16(a, bw, acc);
  }
  int mb = (lane >> 4) * 8, n = lane & 15;
  float bias = b1[f0 + n];
#pragma unroll
  for (int j = 0; j < 8; ++j) {
    int r = r0 + mb + j;
    if (r < Nc) {
      float v = acc[j] + bias;
      if (v < 0.f) v = 0.f;
      h[((long)b * Nc + r) * Fc + f0 + n] = (__bf16)v;
    }
  }
}

// y = x1 + h @ W2.T + b2
__global__ void k_ffn2(const __bf16* __restrict__ h, const __bf16* __restrict__ W2b,
                       const float* __restrict__ b2, const float* __restrict__ x1,
                       float* __restrict__ y) {
  __shared__ __bf16 As[16 * Fc];   // 16 KB
  int b  = blockIdx.x;
  int r0 = blockIdx.y * 16;
  int lane = threadIdx.x & 31;
  int col  = lane & 15;
  int kb   = (lane >> 4) * 16;
  int d0   = (threadIdx.x >> 5) * 16;
  for (int i = threadIdx.x; i < 16 * Fc; i += 256) {
    int r = i >> 9, c = i & (Fc - 1);
    int rr = r0 + r; if (rr >= Nc) rr = Nc - 1;
    As[i] = h[((long)b * Nc + rr) * Fc + c];
  }
  __syncthreads();
  v8f acc = {};
#pragma unroll
  for (int ks = 0; ks < Fc / 32; ++ks) {
    v16bf a; load_a_frag(As + ks * 32, Fc, a);
    v16bf bw = load_b_frag_kcontig(W2b + (long)(d0 + col) * Fc + ks * 32 + kb);
    acc = WMMA_BF16(a, bw, acc);
  }
  int mb = (lane >> 4) * 8, n = lane & 15;
  float bias = b2[d0 + n];
#pragma unroll
  for (int j = 0; j < 8; ++j) {
    int r = r0 + mb + j;
    if (r < Nc) {
      long o = ((long)b * Nc + r) * Dc + d0 + n;
      y[o] = x1[o] + acc[j] + bias;
    }
  }
}

// ---------------- workspace layout (bytes) ----------------
#define XF   ((size_t)Bc * Nc * Dc * 4)    // 16,384,000
#define XBH  ((size_t)Bc * Nc * Dc * 2)    //  8,192,000
#define EKT  ((size_t)Bc * Dc * Np * 2)    //  8,388,608
#define HBH  ((size_t)Bc * Nc * Fc * 2)    // 32,768,000
#define WQN  ((size_t)Lc * Dc * Dc)        // 98,304 elems
#define W1N  ((size_t)Lc * Fc * Dc)        // 393,216 elems
#define O_X    ((size_t)0)
#define O_XB   (O_X + XF)
#define O_SQ   (O_XB + XBH)
#define O_EKT  (O_SQ + XF)
#define O_EKVT (O_EKT + EKT)
#define O_Y    (O_EKVT + EKT)
#define O_X1   (O_Y + XF)
#define O_X1B  (O_X1 + XF)
#define O_H    (O_X1B + XBH)
#define O_WQB  (O_H + HBH)
#define O_WKB  (O_WQB + WQN * 2)
#define O_WVB  (O_WKB + WQN * 2)
#define O_W1B  (O_WVB + WQN * 2)
#define O_W2B  (O_W1B + W1N * 2)
// total ~133.5 MB

extern "C" void kernel_launch(void* const* d_in, const int* in_sizes, int n_in,
                              void* d_out, int out_size, void* d_ws, size_t ws_size,
                              hipStream_t stream) {
  const float* data      = (const float*)d_in[0];
  const float* dist      = (const float*)d_in[1];
  const float* log_scale = (const float*)d_in[2];
  const float* W_emb     = (const float*)d_in[3];
  const float* b_emb     = (const float*)d_in[4];
  const float* W_out     = (const float*)d_in[5];
  const float* b_out     = (const float*)d_in[6];
  const float* Wq        = (const float*)d_in[7];
  const float* Wk        = (const float*)d_in[8];
  const float* Wv        = (const float*)d_in[9];
  const float* g1        = (const float*)d_in[10];
  const float* beta1     = (const float*)d_in[11];
  const float* W1        = (const float*)d_in[12];
  const float* b1        = (const float*)d_in[13];
  const float* W2        = (const float*)d_in[14];
  const float* b2        = (const float*)d_in[15];
  const float* g2        = (const float*)d_in[16];
  const float* beta2     = (const float*)d_in[17];
  const float* alpha     = (const float*)d_in[18];

  char* ws = (char*)d_ws;
  float*  x    = (float*)(ws + O_X);
  __bf16* xb   = (__bf16*)(ws + O_XB);
  float*  sq   = (float*)(ws + O_SQ);
  __bf16* ekT  = (__bf16*)(ws + O_EKT);
  __bf16* ekvT = (__bf16*)(ws + O_EKVT);
  float*  y    = (float*)(ws + O_Y);
  float*  x1   = (float*)(ws + O_X1);
  __bf16* x1b  = (__bf16*)(ws + O_X1B);
  __bf16* h    = (__bf16*)(ws + O_H);
  __bf16* Wqb  = (__bf16*)(ws + O_WQB);
  __bf16* Wkb  = (__bf16*)(ws + O_WKB);
  __bf16* Wvb  = (__bf16*)(ws + O_WVB);
  __bf16* W1b  = (__bf16*)(ws + O_W1B);
  __bf16* W2b  = (__bf16*)(ws + O_W2B);

  // one-time (per launch) weight conversions to bf16
  k_cvt<<<dim3((unsigned)((WQN + 255) / 256)), 256, 0, stream>>>(Wq, Wqb, (long)WQN);
  k_cvt<<<dim3((unsigned)((WQN + 255) / 256)), 256, 0, stream>>>(Wk, Wkb, (long)WQN);
  k_cvt<<<dim3((unsigned)((WQN + 255) / 256)), 256, 0, stream>>>(Wv, Wvb, (long)WQN);
  k_cvt<<<dim3((unsigned)((W1N + 255) / 256)), 256, 0, stream>>>(W1, W1b, (long)W1N);
  k_cvt<<<dim3((unsigned)((W1N + 255) / 256)), 256, 0, stream>>>(W2, W2b, (long)W1N);

  long total = (long)Bc * Nc * Dc;
  k_embed<<<dim3((unsigned)((total + 255) / 256)), 256, 0, stream>>>(data, W_emb, b_emb, x, xb);
  k_reembed<<<dim3(Bc, 2), Dc, 0, stream>>>(W_out, b_out, x, xb);

  for (int l = 0; l < Lc; ++l) {
    k_qkv<<<dim3(Bc, RT), 256, 0, stream>>>(xb, Wqb + (long)l * Dc * Dc,
                                            Wkb + (long)l * Dc * Dc,
                                            Wvb + (long)l * Dc * Dc, sq, ekT, ekvT);
    k_attn<<<dim3(Bc, RT), 256, 0, stream>>>(dist, log_scale, alpha + l, x, sq, ekT, ekvT, y);
    k_inorm<<<dim3(Bc), Dc, 0, stream>>>(y, g1 + (long)l * Dc, beta1 + (long)l * Dc, x1, x1b);
    k_ffn1<<<dim3(Bc, RT, Fc / 128), 256, 0, stream>>>(x1b, W1b + (long)l * Fc * Dc,
                                                       b1 + (long)l * Fc, h);
    k_ffn2<<<dim3(Bc, RT), 256, 0, stream>>>(h, W2b + (long)l * Dc * Fc, b2 + (long)l * Dc, x1, y);
    float* dst = (l == Lc - 1) ? (float*)d_out : x;
    k_inorm<<<dim3(Bc), Dc, 0, stream>>>(y, g2 + (long)l * Dc, beta2 + (long)l * Dc, dst, xb);
  }
}
